// POSITIONLSTM_27066883899583
// MI455X (gfx1250) — matrix operand order repaired and checked by reading, not compiled, not run on hardware
//
#include <hip/hip_runtime.h>
#include <hip/hip_bf16.h>

typedef __attribute__((ext_vector_type(16))) _Float16 v16h;
typedef __attribute__((ext_vector_type(8)))  float    v8f;

#define NUM_FRAMES 50
#define MAX_FRAMES 99
#define WINDOW 10
#define HIDDEN 100
#define LAYERS 5
#define FEAT 4

// WMMA tiling: gates = [10(M, pad16) x K] @ [K x 400(N)]
#define NTILES 25              // 400 / 16
#define KT_MAX 4               // ceil(100/32)
#define TILE_HALFS 512         // 32 lanes * 16 f16 elements per B fragment
#define REGION_HALFS (LAYERS * NTILES * KT_MAX * TILE_HALFS)  // 256000 halfs per region
#define GATE_DUMMY (WINDOW * 400)   // scratch slot for masked-off rows (M >= 10)

// d_ws layout:
//   [0, REGION_HALFS)              f16 : packed Wih B-tiles  (layer0 uses kt=0 only)
//   [REGION_HALFS, 2*REGION_HALFS) f16 : packed Whh B-tiles
//   byte offset 2*REGION_HALFS*2       : float buf1[396], buf2[396]
#define BUF_BYTE_OFF (2 * REGION_HALFS * (int)sizeof(_Float16))

// ---------------------------------------------------------------------------
// Pack fp32 weights into f16 WMMA B-fragment layout (B = W^T, 32x16 per tile).
// B fragment element (lane, e=2v+j):  k = kt*32 + (v<4 ? 2v : 16+2(v-4)) + (lane>>4)*8 + j
//                                     n = nTile*16 + (lane&15)
// value = W[n][k]  (so B[k][n] = W^T[k][n])
// ---------------------------------------------------------------------------
__global__ void pack_wmma_tiles(const float* __restrict__ Wih0,
                                const float* __restrict__ Wih,
                                const float* __restrict__ Whh,
                                _Float16* __restrict__ out) {
    int idx = blockIdx.x * blockDim.x + threadIdx.x;
    if (idx >= 2 * REGION_HALFS) return;
    bool isWhh = idx >= REGION_HALFS;
    int r = isWhh ? idx - REGION_HALFS : idx;
    int ti     = r >> 9;        // tile index
    int within = r & 511;
    int lane = within >> 4;
    int e    = within & 15;
    int v = e >> 1, j = e & 1;
    int l  = ti / (NTILES * KT_MAX);
    int n  = (ti / KT_MAX) % NTILES;
    int kt = ti % KT_MAX;
    int k  = kt * 32 + (v < 4 ? 2 * v : 16 + 2 * (v - 4)) + (lane >> 4) * 8 + j;
    int ng = n * 16 + (lane & 15);      // 0..399
    float val = 0.0f;
    if (!isWhh) {
        if (l == 0) { if (k < FEAT)   val = Wih0[ng * FEAT + k]; }
        else        { if (k < HIDDEN) val = Wih[((l - 1) * 400 + ng) * HIDDEN + k]; }
    } else {
        if (k < HIDDEN) val = Whh[(l * 400 + ng) * HIDDEN + k];
    }
    out[idx] = (_Float16)val;
}

// ---------------------------------------------------------------------------
// Build a 16x32 f16 A fragment from an LDS fp32 source [10 x KDIM].
// A layout (16-bit, 16x32): lane gives M=lane&15, half=lane>>4;
//   elem e=2v+j -> k = KT*32 + (v<4 ? 2v : 16+2(v-4)) + half*8 + j
// ---------------------------------------------------------------------------
template <int KT, int KDIM>
__device__ __forceinline__ v16h build_a(const float* __restrict__ src, int lane) {
    v16h a;
    int m = lane & 15;
    int half8 = (lane >> 4) * 8;
#pragma unroll
    for (int e = 0; e < 16; ++e) {
        const int v = e >> 1, j = e & 1;
        const int kb = (v < 4 ? 2 * v : 16 + 2 * (v - 4));
        int k = KT * 32 + kb + half8 + j;
        float val = (m < WINDOW && k < KDIM) ? src[m * KDIM + k] : 0.0f;
        a[e] = (_Float16)val;
    }
    return a;
}

__device__ inline float sigm(float x) { return 1.0f / (1.0f + __expf(-x)); }

// ---------------------------------------------------------------------------
// One layer's gate GEMM: gates = inp@Wih^T + h@Whh^T + bih + bhh.
// KTIH / KIN are template constants: fragment arrays fully unrolled (plain
// SSA registers, no v_movrels).  Per N-tile: issue ALL B loads first (one
// clause, one L2 round trip), then chain the WMMAs on one accumulator.
// ---------------------------------------------------------------------------
template <int KTIH, int KIN>
__device__ __forceinline__ void layer_gemm(int l, int lane, int wave,
                                           const float* __restrict__ srcIn,
                                           const float* __restrict__ hsrc,
                                           float* __restrict__ gates,
                                           const _Float16* __restrict__ wihPack,
                                           const _Float16* __restrict__ whhPack,
                                           const float* __restrict__ bih,
                                           const float* __restrict__ bhh) {
    v16h aih[KTIH], ahh[KT_MAX];
    if (KTIH >= 1) aih[0] = build_a<0, KIN>(srcIn, lane);
    if (KTIH >= 2) aih[1] = build_a<1, KIN>(srcIn, lane);
    if (KTIH >= 3) aih[2] = build_a<2, KIN>(srcIn, lane);
    if (KTIH >= 4) aih[3] = build_a<3, KIN>(srcIn, lane);
    ahh[0] = build_a<0, HIDDEN>(hsrc, lane);
    ahh[1] = build_a<1, HIDDEN>(hsrc, lane);
    ahh[2] = build_a<2, HIDDEN>(hsrc, lane);
    ahh[3] = build_a<3, HIDDEN>(hsrc, lane);

    for (int n = wave; n < NTILES; n += 8) {
        const int tbase = ((l * NTILES) + n) * KT_MAX;

        // ---- issue every B load up front: all in flight concurrently ----
        v16h bi[KTIH], bh[KT_MAX];
#pragma unroll
        for (int kt = 0; kt < KTIH; ++kt)
            bi[kt] = *(const v16h*)(wihPack + (tbase + kt) * TILE_HALFS + lane * 16);
#pragma unroll
        for (int kt = 0; kt < KT_MAX; ++kt)
            bh[kt] = *(const v16h*)(whhPack + (tbase + kt) * TILE_HALFS + lane * 16);

        const int ngl = n * 16 + (lane & 15);
        const float bias = bih[l * 400 + ngl] + bhh[l * 400 + ngl];

        // ---- back-to-back WMMA accumulation chain ----
        v8f acc = {};
#pragma unroll
        for (int kt = 0; kt < KTIH; ++kt)
            acc = __builtin_amdgcn_wmma_f32_16x16x32_f16(
                false, aih[kt], false, bi[kt], (short)0, acc, false, false);
#pragma unroll
        for (int kt = 0; kt < KT_MAX; ++kt)
            acc = __builtin_amdgcn_wmma_f32_16x16x32_f16(
                false, ahh[kt], false, bh[kt], (short)0, acc, false, false);

        // ---- branchless store: rows M>=10 go to a dummy LDS slot ----
        const int mh = 8 * (lane >> 4);
#pragma unroll
        for (int rI = 0; rI < 8; ++rI) {
            int M = rI + mh;
            int addr = (M < WINDOW) ? (M * 400 + ngl) : (GATE_DUMMY + (lane & 15));
            gates[addr] = acc[rI] + bias;
        }
    }
}

// ---------------------------------------------------------------------------
// Persistent autoregressive LSTM: one block per view, 8 waves.
// ---------------------------------------------------------------------------
__global__ void __launch_bounds__(256)
lstm_autoreg_wmma(const float* __restrict__ x,
                  const _Float16* __restrict__ wsH,     // packed Wih | Whh
                  const float* __restrict__ bih,
                  const float* __restrict__ bhh,
                  const float* __restrict__ lin1W, const float* __restrict__ lin1b,
                  const float* __restrict__ lin2W, const float* __restrict__ lin2b,
                  float* __restrict__ wsBuf) {          // out: 2 x 396 floats
    __shared__ float h_lds[LAYERS * WINDOW * HIDDEN];   // 20000 B
    __shared__ float c_lds[LAYERS * WINDOW * HIDDEN];   // 20000 B
    __shared__ float gates[WINDOW * 400 + 16];          // +16 dummy slot
    __shared__ float buf[MAX_FRAMES * FEAT];            //  1584 B
    __shared__ float p_lds[FEAT];

    const int tid  = threadIdx.x;
    const int lane = tid & 31;
    const int wave = tid >> 5;
    const int view = blockIdx.x;

    const _Float16* wihPack = wsH;
    const _Float16* whhPack = wsH + REGION_HALFS;
    const float* linW = (view == 0) ? lin1W : lin2W;
    const float* linB = (view == 0) ? lin1b : lin2b;

    // Init: frames 0..49 from x[view]; h = c = 0.
    for (int i = tid; i < NUM_FRAMES * FEAT; i += 256)
        buf[i] = x[view * MAX_FRAMES * FEAT + i];
    for (int i = tid; i < LAYERS * WINDOW * HIDDEN; i += 256) {
        h_lds[i] = 0.0f;
        c_lds[i] = 0.0f;
    }
    __syncthreads();

    for (int t = NUM_FRAMES; t < MAX_FRAMES; ++t) {
        for (int l = 0; l < LAYERS; ++l) {
            if (l == 0) {
                layer_gemm<1, FEAT>(0, lane, wave, &buf[(t - WINDOW) * FEAT],
                                    &h_lds[0], gates, wihPack, whhPack, bih, bhh);
            } else {
                layer_gemm<KT_MAX, HIDDEN>(l, lane, wave,
                                           &h_lds[(l - 1) * WINDOW * HIDDEN],
                                           &h_lds[l * WINDOW * HIDDEN],
                                           gates, wihPack, whhPack, bih, bhh);
            }
            __syncthreads();

            // ---- elementwise LSTM cell update (torch gate order i,f,g,o) ----
            for (int idx = tid; idx < WINDOW * HIDDEN; idx += 256) {
                int m = idx / HIDDEN, u = idx % HIDDEN;
                float gi = gates[m * 400 + u];
                float gf = gates[m * 400 + 100 + u];
                float gg = gates[m * 400 + 200 + u];
                float go = gates[m * 400 + 300 + u];
                int s = (l * WINDOW + m) * HIDDEN + u;
                float cn = sigm(gf) * c_lds[s] + sigm(gi) * tanhf(gg);
                float hn = sigm(go) * tanhf(cn);
                c_lds[s] = cn;
                h_lds[s] = hn;
            }
            __syncthreads();
        }

        // ---- linear head: p = flatten(h[4]) @ linW^T + linB  (4 outputs) ----
        if (tid < FEAT) p_lds[tid] = 0.0f;
        __syncthreads();
        {
            const float* o1 = &h_lds[(LAYERS - 1) * WINDOW * HIDDEN];
            int f = tid & 3;
            float part = 0.0f;
            for (int k = tid >> 2; k < WINDOW * HIDDEN; k += 64)
                part += o1[k] * linW[f * (WINDOW * HIDDEN) + k];
            atomicAdd(&p_lds[f], part);
        }
        __syncthreads();
        if (tid < FEAT) buf[t * FEAT + tid] = p_lds[tid] + linB[tid];
        __syncthreads();
    }

    // Export the full 99x4 buffer for the MLP kernel.
    for (int i = tid; i < MAX_FRAMES * FEAT; i += 256)
        wsBuf[view * (MAX_FRAMES * FEAT) + i] = buf[i];
}

// ---------------------------------------------------------------------------
// MLP epilogue: feat[792] -> relu(fc1) -> relu(fc2) -> relu(fc3) -> fc4 -> [2]
// ---------------------------------------------------------------------------
__global__ void __launch_bounds__(256)
mlp_head(const float* __restrict__ wsBuf,
         const float* __restrict__ fc1W, const float* __restrict__ fc1b,
         const float* __restrict__ fc2W, const float* __restrict__ fc2b,
         const float* __restrict__ fc3W, const float* __restrict__ fc3b,
         const float* __restrict__ fc4W, const float* __restrict__ fc4b,
         float* __restrict__ out) {
    __shared__ float feat[792], h1[792], h2[396], h3[198];
    const int tid = threadIdx.x;
    for (int i = tid; i < 792; i += 256) feat[i] = wsBuf[i];
    __syncthreads();
    for (int i = tid; i < 792; i += 256) {
        float s = fc1b[i];
        for (int k = 0; k < 792; ++k) s += fc1W[i * 792 + k] * feat[k];
        h1[i] = fmaxf(s, 0.0f);
    }
    __syncthreads();
    for (int i = tid; i < 396; i += 256) {
        float s = fc2b[i];
        for (int k = 0; k < 792; ++k) s += fc2W[i * 792 + k] * h1[k];
        h2[i] = fmaxf(s, 0.0f);
    }
    __syncthreads();
    for (int i = tid; i < 198; i += 256) {
        float s = fc3b[i];
        for (int k = 0; k < 396; ++k) s += fc3W[i * 396 + k] * h2[k];
        h3[i] = fmaxf(s, 0.0f);
    }
    __syncthreads();
    if (tid < 2) {
        float s = fc4b[tid];
        for (int k = 0; k < 198; ++k) s += fc4W[tid * 198 + k] * h3[k];
        out[tid] = s;
    }
}

extern "C" void kernel_launch(void* const* d_in, const int* in_sizes, int n_in,
                              void* d_out, int out_size, void* d_ws, size_t ws_size,
                              hipStream_t stream) {
    (void)in_sizes; (void)n_in; (void)out_size; (void)ws_size;
    const float* x     = (const float*)d_in[0];
    const float* Wih0  = (const float*)d_in[1];
    const float* Wih   = (const float*)d_in[2];
    const float* Whh   = (const float*)d_in[3];
    const float* bih   = (const float*)d_in[4];
    const float* bhh   = (const float*)d_in[5];
    const float* lin1W = (const float*)d_in[6];
    const float* lin1b = (const float*)d_in[7];
    const float* lin2W = (const float*)d_in[8];
    const float* lin2b = (const float*)d_in[9];
    const float* fc1W  = (const float*)d_in[10];
    const float* fc1b  = (const float*)d_in[11];
    const float* fc2W  = (const float*)d_in[12];
    const float* fc2b  = (const float*)d_in[13];
    const float* fc3W  = (const float*)d_in[14];
    const float* fc3b  = (const float*)d_in[15];
    const float* fc4W  = (const float*)d_in[16];
    const float* fc4b  = (const float*)d_in[17];

    _Float16* wsH = (_Float16*)d_ws;
    float* wsBuf  = (float*)((char*)d_ws + BUF_BYTE_OFF);

    // 1) pack weights into f16 WMMA B-tiles (re-done each call: deterministic)
    int packElems = 2 * REGION_HALFS;
    pack_wmma_tiles<<<(packElems + 255) / 256, 256, 0, stream>>>(Wih0, Wih, Whh, wsH);

    // 2) autoregressive LSTM, one persistent block per view
    lstm_autoreg_wmma<<<2, 256, 0, stream>>>(x, wsH, bih, bhh,
                                             lin1W, lin1b, lin2W, lin2b, wsBuf);

    // 3) MLP head -> d_out[2]
    mlp_head<<<1, 256, 0, stream>>>(wsBuf, fc1W, fc1b, fc2W, fc2b,
                                    fc3W, fc3b, fc4W, fc4b, (float*)d_out);
}
